// SelfBlock_19567871000579
// MI455X (gfx1250) — matrix operand order repaired and checked by reading, not compiled, not run on hardware
//
#include <hip/hip_runtime.h>
#include <hip/hip_bf16.h>

#define NTOK 4096
#define EMB  256
#define NH   4
#define HD   64

typedef __bf16 bf16_t;
typedef bf16_t v16bf __attribute__((ext_vector_type(16)));
typedef bf16_t v8bf  __attribute__((ext_vector_type(8)));
typedef bf16_t v4bf  __attribute__((ext_vector_type(4)));
typedef float  v8f   __attribute__((ext_vector_type(8)));
typedef int    v4i   __attribute__((ext_vector_type(4)));

#define WMMA_BF16(a, b, c) \
  __builtin_amdgcn_wmma_f32_16x16x32_bf16(false, (a), false, (b), (short)0, (c), false, false)

#if __has_builtin(__builtin_amdgcn_global_load_async_to_lds_b128)
#define HAVE_ASYNC_LDS 1
#endif

// 16-byte global -> LDS copy; async (tracked by ASYNCcnt) when available.
__device__ __forceinline__ void copy16_g2l(const bf16_t* g, bf16_t* l) {
#ifdef HAVE_ASYNC_LDS
  __builtin_amdgcn_global_load_async_to_lds_b128(
      (__attribute__((address_space(1))) v4i*)g,
      (__attribute__((address_space(3))) v4i*)l, 0, 0);
#else
  *(v8bf*)l = *(const v8bf*)g;
#endif
}

__device__ __forceinline__ void async_wait0() {
#ifdef HAVE_ASYNC_LDS
#if __has_builtin(__builtin_amdgcn_s_wait_asynccnt)
  __builtin_amdgcn_s_wait_asynccnt(0);
#else
  asm volatile("s_wait_asynccnt 0" ::: "memory");
#endif
#endif
}

// ---------------- fragment loaders ----------------
// A-matrix 16x32 bf16: lane l -> row = l&15, half h = l>>4.
//   j in [0,8):  K = kbase + 8h + j          (contiguous)
//   j in [8,16): K = kbase + 16 + 8h + (j-8) (contiguous)
__device__ __forceinline__ v16bf load_a_bf16(const bf16_t* rowp, int kbase, int half) {
  v8bf lo = *(const v8bf*)(rowp + kbase + half * 8);
  v8bf hi = *(const v8bf*)(rowp + kbase + half * 8 + 16);
  v16bf r;
#pragma unroll
  for (int j = 0; j < 8; ++j) { r[j] = lo[j]; r[j + 8] = hi[j]; }
  return r;
}

// B-matrix 32x16 bf16: lane l -> col = l&15, half h = l>>4; element j<8: K = kbase + 8h + j, j>=8: K = kbase + 16 + 8h + (j-8).
__device__ __forceinline__ v16bf load_b_bf16(const bf16_t* colp, int kbase, int half) {
  v8bf lo = *(const v8bf*)(colp + kbase + half * 8);
  v8bf hi = *(const v8bf*)(colp + kbase + half * 8 + 16);
  v16bf r;
#pragma unroll
  for (int j = 0; j < 8; ++j) { r[j] = lo[j]; r[j + 8] = hi[j]; }
  return r;
}

// ---------------- f32 -> bf16 conversion (n multiple of 4) ----------------
__global__ void cvt_bf16(const float* __restrict__ s, bf16_t* __restrict__ d, int n) {
  int i = (blockIdx.x * blockDim.x + threadIdx.x) * 4;
  if (i >= n) return;
  float4 v = *(const float4*)(s + i);
  v4bf o;
  o[0] = (bf16_t)v.x; o[1] = (bf16_t)v.y; o[2] = (bf16_t)v.z; o[3] = (bf16_t)v.w;
  *(v4bf*)(d + i) = o;
}

// ---------------- GEMM: C = A @ W^T + bias (+res), A/W in bf16 ----------------
// A: MxK bf16 (lda), W: NoutxK bf16 (ldw), C: MxNout f32 (ldc).
// grid = (M/64, Nout/64), block = 128 (4 waves). Wave w: rows m0+16w..+16, all 64 cols.
// B tile (64 cols x 32 K, 4KB) staged in LDS via async loads, reused by all 4 waves.
__global__ void gemm_bias(const bf16_t* __restrict__ A, int lda,
                          const bf16_t* __restrict__ W, int ldw,
                          const float* __restrict__ bias,
                          const float* __restrict__ res, int ldr,
                          float* __restrict__ C, int ldc, int K) {
  __shared__ bf16_t bs[64][32];
  const int tid = threadIdx.x;
  const int lane = tid & 31;
  const int wave = tid >> 5;
  const int m0 = blockIdx.x * 64 + wave * 16;
  const int n0 = blockIdx.y * 64;
  const int half = lane >> 4;
  const int lr = lane & 15;
  const bf16_t* arow = A + (size_t)(m0 + lr) * lda;

  v8f acc[4] = {};
  for (int k = 0; k < K; k += 32) {
    // stage B tile: 256 x 16B chunks, 2 per thread
#pragma unroll
    for (int u = 0; u < 2; ++u) {
      int idx = tid * 2 + u;
      int rrow = idx >> 2, chunk = idx & 3;
      copy16_g2l(W + (size_t)(n0 + rrow) * ldw + k + chunk * 8, &bs[rrow][chunk * 8]);
    }
    async_wait0();
    __syncthreads();

    v16bf a = load_a_bf16(arow, k, half);
#pragma unroll
    for (int t = 0; t < 4; ++t) {
      v16bf b = load_b_bf16(&bs[t * 16 + lr][0], 0, half);
      acc[t] = WMMA_BF16(a, b, acc[t]);
    }
    __syncthreads();
  }

  if (res) {
#pragma unroll
    for (int t = 0; t < 4; ++t) {
      int col = n0 + t * 16 + lr;
      float bv = bias[col];
#pragma unroll
      for (int i = 0; i < 8; ++i) {
        int row = m0 + half * 8 + i;
        C[(size_t)row * ldc + col] = acc[t][i] + bv + res[(size_t)row * ldr + col];
      }
    }
  } else {
#pragma unroll
    for (int t = 0; t < 4; ++t) {
      int col = n0 + t * 16 + lr;
      float bv = bias[col];
#pragma unroll
      for (int i = 0; i < 8; ++i) {
        int row = m0 + half * 8 + i;
        C[(size_t)row * ldc + col] = acc[t][i] + bv;
      }
    }
  }
}

// ---------------- rotary + bf16 repack ----------------
// qkv: (N, 768), last dim = (H, D, 3). qb/kb: (H,N,D) bf16 (q pre-scaled by 1/8);
// vt: (H,D,N) bf16 transposed so P@V B-fragments are contiguous.
__global__ void rotary_pack(const float* __restrict__ qkv, const float* __restrict__ enc,
                            bf16_t* __restrict__ qb, bf16_t* __restrict__ kb,
                            bf16_t* __restrict__ vt) {
  int idx = blockIdx.x * blockDim.x + threadIdx.x;
  if (idx >= NH * NTOK * (HD / 2)) return;
  int p = idx & 31;
  int n = (idx >> 5) & (NTOK - 1);
  int h = idx >> 17;
  int d0 = 2 * p, d1 = 2 * p + 1;
  const float* base = qkv + (size_t)n * 768 + h * 192;
  float q0 = base[d0 * 3 + 0], q1 = base[d1 * 3 + 0];
  float k0 = base[d0 * 3 + 1], k1 = base[d1 * 3 + 1];
  float v0 = base[d0 * 3 + 2], v1 = base[d1 * 3 + 2];
  const float* cosp = enc + (size_t)n * HD;
  const float* sinp = enc + (size_t)NTOK * HD + (size_t)n * HD;
  float c0 = cosp[d0], c1 = cosp[d1], s0 = sinp[d0], s1 = sinp[d1];
  size_t ro = ((size_t)h * NTOK + n) * HD;
  qb[ro + d0] = (bf16_t)((q0 * c0 - q1 * s0) * 0.125f);
  qb[ro + d1] = (bf16_t)((q1 * c1 + q0 * s1) * 0.125f);
  kb[ro + d0] = (bf16_t)(k0 * c0 - k1 * s0);
  kb[ro + d1] = (bf16_t)(k1 * c1 + k0 * s1);
  vt[((size_t)h * HD + d0) * NTOK + n] = (bf16_t)v0;
  vt[((size_t)h * HD + d1) * NTOK + n] = (bf16_t)v1;
}

// ---------------- flash attention ----------------
// grid = (N/64, H), block = 128 (4 waves); wave owns 16 query rows; ctx out is bf16.
__global__ void flash_attn(const bf16_t* __restrict__ qb, const bf16_t* __restrict__ kb,
                           const bf16_t* __restrict__ vt, bf16_t* __restrict__ ctxb) {
  __shared__ bf16_t pbuf[4][16][32];
  const int lane = threadIdx.x & 31;
  const int wave = threadIdx.x >> 5;
  const int head = blockIdx.y;
  const int q0 = blockIdx.x * 64 + wave * 16;
  const int half = lane >> 4;
  const int lr = lane & 15;

  const bf16_t* qrow = qb + ((size_t)head * NTOK + q0 + lr) * HD;
  v16bf qlo = load_a_bf16(qrow, 0, half);
  v16bf qhi = load_a_bf16(qrow, 32, half);

  float mrow[8], lrow[8];
  v8f acc[4] = {};
#pragma unroll
  for (int i = 0; i < 8; ++i) { mrow[i] = -3.0e38f; lrow[i] = 0.f; }

  for (int kt = 0; kt < NTOK; kt += 32) {
    v8f s0 = {}, s1 = {};
    const bf16_t* kcol0 = kb + ((size_t)head * NTOK + kt + lr) * HD;
    const bf16_t* kcol1 = kcol0 + 16 * HD;
    s0 = WMMA_BF16(qlo, load_b_bf16(kcol0, 0, half), s0);
    s0 = WMMA_BF16(qhi, load_b_bf16(kcol0, 32, half), s0);
    s1 = WMMA_BF16(qlo, load_b_bf16(kcol1, 0, half), s1);
    s1 = WMMA_BF16(qhi, load_b_bf16(kcol1, 32, half), s1);

#pragma unroll
    for (int i = 0; i < 8; ++i) {
      float a0 = s0[i], a1 = s1[i];
      float mx = fmaxf(a0, a1);
#pragma unroll
      for (int d = 1; d < 16; d <<= 1) mx = fmaxf(mx, __shfl_xor(mx, d, 32));
      float mnew = fmaxf(mrow[i], mx);
      float alpha = __expf(mrow[i] - mnew);
      float p0 = __expf(a0 - mnew);
      float p1 = __expf(a1 - mnew);
      float rs = p0 + p1;
#pragma unroll
      for (int d = 1; d < 16; d <<= 1) rs += __shfl_xor(rs, d, 32);
      lrow[i] = lrow[i] * alpha + rs;
      mrow[i] = mnew;
#pragma unroll
      for (int t = 0; t < 4; ++t) acc[t][i] *= alpha;
      pbuf[wave][half * 8 + i][lr]      = (bf16_t)p0;
      pbuf[wave][half * 8 + i][16 + lr] = (bf16_t)p1;
    }

    v16bf pf = load_a_bf16(&pbuf[wave][lr][0], 0, half);
#pragma unroll
    for (int t = 0; t < 4; ++t) {
      const bf16_t* vcol = vt + ((size_t)head * HD + t * 16 + lr) * NTOK + kt;
      acc[t] = WMMA_BF16(pf, load_b_bf16(vcol, 0, half), acc[t]);
    }
  }

#pragma unroll
  for (int i = 0; i < 8; ++i) {
    float inv = 1.0f / lrow[i];
    int row = q0 + half * 8 + i;
#pragma unroll
    for (int t = 0; t < 4; ++t)
      ctxb[(size_t)row * EMB + head * HD + t * 16 + lr] = (bf16_t)(acc[t][i] * inv);
  }
}

// ---------------- concat (x, message) -> bf16 ----------------
__global__ void concat_xm(const float* __restrict__ x, const float* __restrict__ m,
                          bf16_t* __restrict__ h) {
  int i = blockIdx.x * blockDim.x + threadIdx.x;
  if (i >= NTOK * 512) return;
  int n = i >> 9, j = i & 511;
  h[i] = (bf16_t)((j < 256) ? x[n * 256 + j] : m[n * 256 + (j - 256)]);
}

// ---------------- LayerNorm + exact GELU -> bf16 (one block per row of 512) ----------------
__global__ void ln_gelu(const float* __restrict__ in, const float* __restrict__ g,
                        const float* __restrict__ b, bf16_t* __restrict__ outp) {
  const int row = blockIdx.x;
  const int tid = threadIdx.x;
  const float* rp = in + (size_t)row * 512;
  float x0 = rp[tid], x1 = rp[tid + 256];
  __shared__ float sred[8];
  float s = x0 + x1;
#pragma unroll
  for (int d = 1; d < 32; d <<= 1) s += __shfl_xor(s, d, 32);
  if ((tid & 31) == 0) sred[tid >> 5] = s;
  __syncthreads();
  float tot = 0.f;
#pragma unroll
  for (int i = 0; i < 8; ++i) tot += sred[i];
  float mu = tot * (1.0f / 512.0f);
  __syncthreads();
  float d0 = x0 - mu, d1 = x1 - mu;
  float v = d0 * d0 + d1 * d1;
#pragma unroll
  for (int d = 1; d < 32; d <<= 1) v += __shfl_xor(v, d, 32);
  if ((tid & 31) == 0) sred[tid >> 5] = v;
  __syncthreads();
  float vtot = 0.f;
#pragma unroll
  for (int i = 0; i < 8; ++i) vtot += sred[i];
  float rstd = rsqrtf(vtot * (1.0f / 512.0f) + 1e-5f);
  float y0 = d0 * rstd * g[tid] + b[tid];
  float y1 = d1 * rstd * g[tid + 256] + b[tid + 256];
  y0 = 0.5f * y0 * (1.0f + erff(y0 * 0.70710678118654752f));
  y1 = 0.5f * y1 * (1.0f + erff(y1 * 0.70710678118654752f));
  outp[(size_t)row * 512 + tid] = (bf16_t)y0;
  outp[(size_t)row * 512 + tid + 256] = (bf16_t)y1;
}

extern "C" void kernel_launch(void* const* d_in, const int* in_sizes, int n_in,
                              void* d_out, int out_size, void* d_ws, size_t ws_size,
                              hipStream_t stream) {
  const float* x    = (const float*)d_in[0];
  const float* enc  = (const float*)d_in[1];
  const float* Wqkv = (const float*)d_in[2];
  const float* bqkv = (const float*)d_in[3];
  const float* Wout = (const float*)d_in[4];
  const float* bout = (const float*)d_in[5];
  const float* W1   = (const float*)d_in[6];
  const float* b1   = (const float*)d_in[7];
  const float* lng  = (const float*)d_in[8];
  const float* lnb  = (const float*)d_in[9];
  const float* W2   = (const float*)d_in[10];
  const float* b2   = (const float*)d_in[11];
  float* out = (float*)d_out;

  char* ws = (char*)d_ws;
  size_t off = 0;
  auto alloc = [&](size_t bytes) -> void* {
    void* p = ws + off;
    off += (bytes + 255) & ~(size_t)255;
    return p;
  };
  float*  qkv   = (float*)alloc((size_t)NTOK * 768 * 4);
  bf16_t* qb    = (bf16_t*)alloc((size_t)NH * NTOK * HD * 2);
  bf16_t* kbuf  = (bf16_t*)alloc((size_t)NH * NTOK * HD * 2);
  bf16_t* vt    = (bf16_t*)alloc((size_t)NH * NTOK * HD * 2);
  bf16_t* ctxb  = (bf16_t*)alloc((size_t)NTOK * EMB * 2);
  float*  msg   = (float*)alloc((size_t)NTOK * EMB * 4);
  bf16_t* hb    = (bf16_t*)alloc((size_t)NTOK * 512 * 2);
  float*  h1    = (float*)alloc((size_t)NTOK * 512 * 4);
  bf16_t* h2b   = (bf16_t*)alloc((size_t)NTOK * 512 * 2);
  bf16_t* xb    = (bf16_t*)alloc((size_t)NTOK * EMB * 2);
  bf16_t* Wqkvb = (bf16_t*)alloc((size_t)768 * EMB * 2);
  bf16_t* Woutb = (bf16_t*)alloc((size_t)EMB * EMB * 2);
  bf16_t* W1b   = (bf16_t*)alloc((size_t)512 * 512 * 2);
  bf16_t* W2b   = (bf16_t*)alloc((size_t)EMB * 512 * 2);

  // 0) one-time bf16 conversions of GEMM operands
  cvt_bf16<<<(NTOK * EMB / 4 + 255) / 256, 256, 0, stream>>>(x, xb, NTOK * EMB);
  cvt_bf16<<<(768 * EMB / 4 + 255) / 256, 256, 0, stream>>>(Wqkv, Wqkvb, 768 * EMB);
  cvt_bf16<<<(EMB * EMB / 4 + 255) / 256, 256, 0, stream>>>(Wout, Woutb, EMB * EMB);
  cvt_bf16<<<(512 * 512 / 4 + 255) / 256, 256, 0, stream>>>(W1, W1b, 512 * 512);
  cvt_bf16<<<(EMB * 512 / 4 + 255) / 256, 256, 0, stream>>>(W2, W2b, EMB * 512);

  dim3 blk(128);
  // 1) QKV projection
  gemm_bias<<<dim3(NTOK / 64, 768 / 64), blk, 0, stream>>>(xb, EMB, Wqkvb, EMB, bqkv,
                                                           nullptr, 0, qkv, 768, EMB);
  // 2) rotary + bf16 repack (q pre-scaled by 1/sqrt(D))
  rotary_pack<<<(NH * NTOK * (HD / 2) + 255) / 256, 256, 0, stream>>>(qkv, enc, qb, kbuf, vt);
  // 3) flash attention -> ctx (N, E) bf16
  flash_attn<<<dim3(NTOK / 64, NH), blk, 0, stream>>>(qb, kbuf, vt, ctxb);
  // 4) out projection
  gemm_bias<<<dim3(NTOK / 64, EMB / 64), blk, 0, stream>>>(ctxb, EMB, Woutb, EMB, bout,
                                                           nullptr, 0, msg, EMB, EMB);
  // 5) concat(x, message) -> bf16
  concat_xm<<<(NTOK * 512 + 255) / 256, 256, 0, stream>>>(x, msg, hb);
  // 6) FFN1
  gemm_bias<<<dim3(NTOK / 64, 512 / 64), blk, 0, stream>>>(hb, 512, W1b, 512, b1,
                                                           nullptr, 0, h1, 512, 512);
  // 7) LayerNorm + GELU -> bf16
  ln_gelu<<<NTOK, 256, 0, stream>>>(h1, lng, lnb, h2b);
  // 8) FFN2 + residual x -> out
  gemm_bias<<<dim3(NTOK / 64, EMB / 64), blk, 0, stream>>>(h2b, 512, W2b, 512, b2,
                                                           x, EMB, out, EMB, 512);
}
